// IARTv2_31696858644755
// MI455X (gfx1250) — matrix operand order repaired and checked by reading, not compiled, not run on hardware
//
#include <hip/hip_runtime.h>
#include <hip/hip_bf16.h>

typedef __attribute__((ext_vector_type(16))) _Float16 v16h;
typedef __attribute__((ext_vector_type(8)))  _Float16 v8h;
typedef __attribute__((ext_vector_type(4)))  _Float16 v4h;
typedef __attribute__((ext_vector_type(8)))  float    v8f;
typedef __attribute__((ext_vector_type(4)))  unsigned int u32x4;
typedef __attribute__((ext_vector_type(8)))  int      i32x8;
typedef __attribute__((ext_vector_type(4)))  int      i32x4;

#define EMBED 80
#define HEADS 4
#define DH 20
#define HH 64
#define WW 64
#define HWX (HH*WW)          // 4096
#define FS  (EMBED*HWX)      // one feature frame, 327680 floats
#define FLS (2*HWX)          // one flow frame
#define TOK3 (3*HWX)         // 12288 tokens in backbone
#define TOKP (4*HWX)         // 16384 padded (dp=4) window tokens
#define ATT_SCALE 0.2236067977f   // 1/sqrt(20)

#if __has_builtin(__builtin_amdgcn_tensor_load_to_lds) && \
    __has_builtin(__builtin_amdgcn_s_wait_tensorcnt)
#define USE_TDM 1
#else
#define USE_TDM 0
#endif

#if defined(__AMDGCN__) && !USE_TDM
#warning "gfx1250 TDM builtins unavailable on device pass: manual B staging in use"
#endif

#if USE_TDM
// 2D TDM descriptor: f16 row-major panel, row length `rowlen` halves,
// `rows_tensor` valid rows (OOB rows read as zero), 64-row tile into LDS.
__device__ __forceinline__ void tdm_load_panel(void* lds, const void* gptr,
                                               int rowlen, int rows_tensor)
{
  unsigned long long ga = (unsigned long long)(size_t)gptr;
  u32x4 g0;
  g0[0] = 1u;                                   // count=1, user descriptor
  g0[1] = (unsigned)(size_t)lds;                // lds_addr
  g0[2] = (unsigned)ga;                         // global_addr[31:0]
  g0[3] = (unsigned)((ga >> 32) & 0x01FFFFFFull) | (2u << 30); // addr[56:32], type=2
  i32x8 g1;
  g1[0] = 0x00010000;                           // workgroup_mask=0, data_size=1 (2B)
  g1[1] = (rowlen & 0xFFFF) << 16;              // tensor_dim0[15:0]
  g1[2] = ((rowlen >> 16) & 0xFFFF) | ((rows_tensor & 0xFFFF) << 16); // td0 hi | td1 lo
  g1[3] = ((rows_tensor >> 16) & 0xFFFF) | ((rowlen & 0xFFFF) << 16); // td1 hi | tile_dim0
  g1[4] = 64;                                   // tile_dim1=64, tile_dim2=0
  g1[5] = rowlen;                               // tensor_dim0_stride[31:0]
  g1[6] = 0;
  g1[7] = 0;
  i32x4 z4 = {0, 0, 0, 0};
#if __clang_major__ >= 23
  i32x8 z8 = {0, 0, 0, 0, 0, 0, 0, 0};
  __builtin_amdgcn_tensor_load_to_lds(g0, g1, z4, z4, z8, 0);
#else
  __builtin_amdgcn_tensor_load_to_lds(g0, g1, z4, z4, 0);
#endif
  __builtin_amdgcn_s_wait_tensorcnt(0);
}
#endif

// ---------------------------------------------------------------------------
// WMMA GEMM:  Y[M,N] = act( X[M,K](f32) * W[N,K](f16)^T + bias + res )
// M multiple of 64 (all call sites). K <= 320, K % 16 == 0.
// Block = 256 threads (8 waves), tile 64(M) x 64(N):
//   wave (wm,wn) owns two 16x16 tiles stacked in M, sharing one B fragment.
// Full K chunks are unguarded (vectorizable LDS traffic); one guarded tail.
// ---------------------------------------------------------------------------
__global__ __launch_bounds__(256) void k_gemm(
    const float* __restrict__ X, const _Float16* __restrict__ W,
    const float* __restrict__ bias, const float* __restrict__ res,
    float* __restrict__ Y, int M, int N, int K, int act)
{
  __shared__ _Float16 As[64][40];
  __shared__ _Float16 Bs[64 * 320];
  const int m0 = blockIdx.y * 64, n0 = blockIdx.x * 64;
  const int tid = threadIdx.x;
  const int wv = tid >> 5, l = tid & 31, g = l >> 4, ln = l & 15;
  const int wm = wv >> 2, wn = wv & 3;

  // ---- load the whole B panel once
#if USE_TDM
  if (tid < 32) tdm_load_panel(&Bs[0], W + (size_t)n0 * K, K, N - n0);
#else
  for (int i = tid; i < 64 * K; i += 256) {
    int nn = n0 + i / K, kk = i % K;
    Bs[i] = (nn < N) ? W[(size_t)nn * K + kk] : (_Float16)0.f;
  }
#endif
  __syncthreads();

  const int ar = tid >> 2;            // 0..63 : A row
  const int ak = (tid & 3) * 8;       // 0..24 : 8 consecutive K halves
  const size_t arow = (size_t)(m0 + ar) * K;
  const int brow = (wn * 16 + ln) * K;
  const int amr = wm * 32 + ln;       // first A-fragment LDS row

  v8f acc0 = {}, acc1 = {};
  const int kfull = K & ~31;
  for (int k0 = 0; k0 < kfull; k0 += 32) {
    v8h pk;
#pragma unroll
    for (int j = 0; j < 8; ++j) pk[j] = (_Float16)X[arow + k0 + ak + j];
    *(v8h*)&As[ar][ak] = pk;
    if (k0 + 32 < K) __builtin_prefetch(&X[arow + k0 + 32], 0, 1);
    __syncthreads();
    v16h a0, a1, b;
#pragma unroll
    for (int e = 0; e < 16; ++e) {
      int ka = g * 16 + e;
      a0[e] = As[amr][ka];
      a1[e] = As[amr + 16][ka];
      b[e]  = Bs[brow + k0 + g * 16 + e];
    }
    acc0 = __builtin_amdgcn_wmma_f32_16x16x32_f16(false, a0, false, b,
                                                  (short)0, acc0, false, false);
    acc1 = __builtin_amdgcn_wmma_f32_16x16x32_f16(false, a1, false, b,
                                                  (short)0, acc1, false, false);
    __syncthreads();
  }
  if (kfull < K) {                     // guarded tail chunk (K % 32 == 16 here)
    v8h pk;
#pragma unroll
    for (int j = 0; j < 8; ++j) {
      int kg = kfull + ak + j;
      int kgc = kg < K ? kg : K - 1;
      float v = X[arow + kgc];
      pk[j] = (_Float16)(kg < K ? v : 0.f);
    }
    *(v8h*)&As[ar][ak] = pk;
    __syncthreads();
    v16h a0, a1, b;
#pragma unroll
    for (int e = 0; e < 16; ++e) {
      int ka = g * 16 + e;
      a0[e] = As[amr][ka];
      a1[e] = As[amr + 16][ka];
      int kg = kfull + g * 16 + e;
      int kgc = kg < K ? kg : 0;
      _Float16 bv = Bs[brow + kgc];
      b[e] = (kg < K) ? bv : (_Float16)0.f;
    }
    acc0 = __builtin_amdgcn_wmma_f32_16x16x32_f16(false, a0, false, b,
                                                  (short)0, acc0, false, false);
    acc1 = __builtin_amdgcn_wmma_f32_16x16x32_f16(false, a1, false, b,
                                                  (short)0, acc1, false, false);
  }
#pragma unroll
  for (int r = 0; r < 8; ++r) {
    int mA = m0 + wm * 32 + r + 8 * g;
    int n = n0 + wn * 16 + ln;
    if (n < N) {
      float v0 = acc0[r], v1 = acc1[r];
      if (bias) { float bb = bias[n]; v0 += bb; v1 += bb; }
      if (res)  { v0 += res[(size_t)mA * N + n]; v1 += res[(size_t)(mA + 16) * N + n]; }
      if (act)  { v0 = (v0 >= 0.f) ? v0 : 0.1f * v0;
                  v1 = (v1 >= 0.f) ? v1 : 0.1f * v1; }
      Y[(size_t)mA * N + n] = v0;
      Y[(size_t)(mA + 16) * N + n] = v1;
    }
  }
}

// ---------------------------------------------------------------------------
// Implicit-GEMM conv on WMMA. x: NCHW f32. wpk: f16 packed [tap][Co][Ci].
// y: NCHW. M = NB*H*W (multiple of 64), N = Co, K = Ci per tap.
// Tile 64x64, two accumulators per wave. B panel per tap via TDM.
// ---------------------------------------------------------------------------
__global__ __launch_bounds__(256) void k_conv_gemm(
    const float* __restrict__ x, const _Float16* __restrict__ wpk,
    const float* __restrict__ bias, float* __restrict__ y,
    int NB, int Ci, int Co, int H, int W, int Kk, int pad, int act)
{
  __shared__ _Float16 As[64][40];
  __shared__ _Float16 Bs[64 * 80];
  const int m0 = blockIdx.y * 64, n0 = blockIdx.x * 64;
  const int tid = threadIdx.x;
  const int wv = tid >> 5, l = tid & 31, g = l >> 4, ln = l & 15;
  const int wm = wv >> 2, wn = wv & 3;

  const int ar = tid >> 2;
  const int ak = (tid & 3) * 8;
  const int am = m0 + ar;
  const int axx = am % W, ayy = (am / W) % H, anb = am / (W * H);
  const int brow = (wn * 16 + ln) * Ci;
  const int amr = wm * 32 + ln;
  const int nchunk = (Ci + 31) >> 5;

  v8f acc0 = {}, acc1 = {};
  for (int tap = 0; tap < Kk * Kk; ++tap) {
    int ky = tap / Kk, kx = tap % Kk;
    __syncthreads();                       // previous tap fully consumed
#if USE_TDM
    if (tid < 32)
      tdm_load_panel(&Bs[0], wpk + ((size_t)tap * Co + n0) * Ci, Ci, Co - n0);
#else
    for (int i = tid; i < 64 * Ci; i += 256) {
      int nn = n0 + i / Ci, kk = i % Ci;
      Bs[i] = (nn < Co) ? wpk[((size_t)tap * Co + nn) * Ci + kk] : (_Float16)0.f;
    }
#endif
    __syncthreads();

    int sy = ayy + ky - pad, sx = axx + kx - pad;
    bool sok = (sy >= 0 && sy < H && sx >= 0 && sx < W);
    int syc = sy < 0 ? 0 : (sy > H - 1 ? H - 1 : sy);
    int sxc = sx < 0 ? 0 : (sx > W - 1 ? W - 1 : sx);
    const size_t xbase = ((size_t)anb * Ci) * H * W + (size_t)syc * W + sxc;

    for (int kc = 0; kc < nchunk; ++kc) {
      v8h pk;
#pragma unroll
      for (int j = 0; j < 8; ++j) {
        int ci = kc * 32 + ak + j;
        int cic = ci < Ci ? ci : Ci - 1;
        float v = x[xbase + (size_t)cic * H * W];
        pk[j] = (_Float16)((sok && ci < Ci) ? v : 0.f);
      }
      *(v8h*)&As[ar][ak] = pk;
      __syncthreads();
      v16h a0, a1, b;
      if (kc * 32 + 32 <= Ci) {            // full chunk: unguarded B fragment
#pragma unroll
        for (int e = 0; e < 16; ++e) {
          int ka = g * 16 + e;
          a0[e] = As[amr][ka];
          a1[e] = As[amr + 16][ka];
          b[e]  = Bs[brow + kc * 32 + g * 16 + e];
        }
      } else {                             // tail chunk
#pragma unroll
        for (int e = 0; e < 16; ++e) {
          int ka = g * 16 + e;
          a0[e] = As[amr][ka];
          a1[e] = As[amr + 16][ka];
          int kg = kc * 32 + g * 16 + e;
          int kgc = kg < Ci ? kg : 0;
          _Float16 bv = Bs[brow + kgc];
          b[e] = (kg < Ci) ? bv : (_Float16)0.f;
        }
      }
      acc0 = __builtin_amdgcn_wmma_f32_16x16x32_f16(false, a0, false, b,
                                                    (short)0, acc0, false, false);
      acc1 = __builtin_amdgcn_wmma_f32_16x16x32_f16(false, a1, false, b,
                                                    (short)0, acc1, false, false);
      __syncthreads();
    }
  }
#pragma unroll
  for (int r = 0; r < 8; ++r) {
    int n = n0 + wn * 16 + ln;
    if (n < Co) {
      int mA = m0 + wm * 32 + r + 8 * g;
      int mB = mA + 16;
      float v0 = acc0[r] + bias[n];
      float v1 = acc1[r] + bias[n];
      if (act) { v0 = (v0 >= 0.f) ? v0 : 0.1f * v0;
                 v1 = (v1 >= 0.f) ? v1 : 0.1f * v1; }
      {
        int xx = mA % W, yy = (mA / W) % H, nb = mA / (W * H);
        y[(((size_t)nb * Co + n) * H + yy) * W + xx] = v0;
      }
      {
        int xx = mB % W, yy = (mB / W) % H, nb = mB / (W * H);
        y[(((size_t)nb * Co + n) * H + yy) * W + xx] = v1;
      }
    }
  }
}

// ---------------------------------------------------------------------------
// Window attention: one block per (window, head). L=128, dh=20 (padded to 32).
// ---------------------------------------------------------------------------
__global__ __launch_bounds__(256) void k_win_attn(
    const float* __restrict__ qkv, float* __restrict__ out, int nWin)
{
  const int wi = blockIdx.x / HEADS;
  const int hd = blockIdx.x % HEADS;
  if (wi >= nWin) return;
  __shared__ _Float16 sQ[128][40];
  __shared__ _Float16 sK[128][40];
  __shared__ _Float16 sV[128][40];
  __shared__ _Float16 sP[128][132];

  const int tid = threadIdx.x;
  const int wv = tid >> 5, l = tid & 31, g = l >> 4, ln = l & 15;
  const int m0 = wv * 16;
  const float* base = qkv + (size_t)wi * 128 * (3 * EMBED);

  for (int i = tid; i < 128 * 32; i += 256) {
    int r = i >> 5, d = i & 31;
    float qv = 0.f, kv = 0.f, vv = 0.f;
    if (d < DH) {
      size_t ro = (size_t)r * 3 * EMBED + hd * DH + d;
      qv = base[ro];
      kv = base[ro + EMBED];
      vv = base[ro + 2 * EMBED];
    }
    sQ[r][d] = (_Float16)qv;
    sK[r][d] = (_Float16)kv;
    sV[r][d] = (_Float16)vv;
  }
  __syncthreads();

  v16h a;
#pragma unroll
  for (int e = 0; e < 16; ++e) {
    int ka = g * 16 + e;
    a[e] = sQ[m0 + ln][ka];
  }
  v8f acc[8];
#pragma unroll
  for (int t = 0; t < 8; ++t) {
    v16h b;
#pragma unroll
    for (int e = 0; e < 16; ++e) b[e] = sK[t * 16 + ln][g * 16 + e];
    v8f z = {};
    acc[t] = __builtin_amdgcn_wmma_f32_16x16x32_f16(false, a, false, b,
                                                    (short)0, z, false, false);
  }
#pragma unroll
  for (int r = 0; r < 8; ++r) {
    float mx = -3.4e38f;
#pragma unroll
    for (int t = 0; t < 8; ++t) mx = fmaxf(mx, acc[t][r] * ATT_SCALE);
    for (int o = 1; o < 16; o <<= 1) mx = fmaxf(mx, __shfl_xor(mx, o, 16));
    float s = 0.f;
#pragma unroll
    for (int t = 0; t < 8; ++t) {
      float e = __expf(acc[t][r] * ATT_SCALE - mx);
      acc[t][r] = e;
      s += e;
    }
    for (int o = 1; o < 16; o <<= 1) s += __shfl_xor(s, o, 16);
    float inv = 1.f / s;
#pragma unroll
    for (int t = 0; t < 8; ++t)
      sP[m0 + r + 8 * g][t * 16 + ln] = (_Float16)(acc[t][r] * inv);
  }
  __syncthreads();

  v8f oacc0 = {}, oacc1 = {};
#pragma unroll
  for (int kc = 0; kc < 4; ++kc) {
    v16h pa;
#pragma unroll
    for (int e = 0; e < 16; ++e) {
      int ka = kc * 32 + g * 16 + e;
      pa[e] = sP[m0 + ln][ka];
    }
    v16h vb0, vb1;
#pragma unroll
    for (int e = 0; e < 16; ++e) {
      vb0[e] = sV[kc * 32 + g * 16 + e][ln];
      vb1[e] = sV[kc * 32 + g * 16 + e][16 + ln];
    }
    oacc0 = __builtin_amdgcn_wmma_f32_16x16x32_f16(false, pa, false, vb0,
                                                   (short)0, oacc0, false, false);
    oacc1 = __builtin_amdgcn_wmma_f32_16x16x32_f16(false, pa, false, vb1,
                                                   (short)0, oacc1, false, false);
  }
#pragma unroll
  for (int r = 0; r < 8; ++r) {
    int m = m0 + r + 8 * g;
    size_t ro = ((size_t)wi * 128 + m) * EMBED + hd * DH;
    if (ln < DH) out[ro + ln] = oacc0[r];
    if (16 + ln < DH) out[ro + 16 + ln] = oacc1[r];
  }
}

// ---------------------------------------------------------------------------
// weight packing (f32 -> f16 panels), run once per launch
// ---------------------------------------------------------------------------
__global__ void k_pack_w(const float* __restrict__ in, _Float16* __restrict__ out, int n)
{
  int i = blockIdx.x * blockDim.x + threadIdx.x;
  if (i < n) out[i] = (_Float16)in[i];
}
// in [K,N] -> out [N,K]
__global__ void k_pack_wT(const float* __restrict__ in, _Float16* __restrict__ out,
                          int N, int K)
{
  int i = blockIdx.x * blockDim.x + threadIdx.x;
  if (i >= N * K) return;
  int n = i / K, k = i % K;
  out[i] = (_Float16)in[(size_t)k * N + n];
}
// in [Co,Ci,Kk,Kk] -> out [tap][Co][Ci]
__global__ void k_pack_conv(const float* __restrict__ in, _Float16* __restrict__ out,
                            int Co, int Ci, int Kk)
{
  int i = blockIdx.x * blockDim.x + threadIdx.x;
  int total = Kk * Kk * Co * Ci;
  if (i >= total) return;
  int ci = i % Ci, co = (i / Ci) % Co, tap = i / (Ci * Co);
  int ky = tap / Kk, kx = tap % Kk;
  out[i] = (_Float16)in[(((size_t)co * Ci + ci) * Kk + ky) * Kk + kx];
}

// ---------------------------------------------------------------------------
// elementwise / glue kernels
// ---------------------------------------------------------------------------
__global__ void k_layernorm(const float* __restrict__ x, const float* __restrict__ g,
                            const float* __restrict__ b, float* __restrict__ y, int T)
{
  int t = blockIdx.x * blockDim.x + threadIdx.x;
  if (t >= T) return;
  const float* xr = x + (size_t)t * EMBED;
  float mu = 0.f;
  for (int c = 0; c < EMBED; ++c) mu += xr[c];
  mu *= (1.f / EMBED);
  float var = 0.f;
  for (int c = 0; c < EMBED; ++c) { float d = xr[c] - mu; var += d * d; }
  var *= (1.f / EMBED);
  float rs = rsqrtf(var + 1e-5f);
  float* yr = y + (size_t)t * EMBED;
  for (int c = 0; c < EMBED; ++c) yr[c] = (xr[c] - mu) * rs * g[c] + b[c];
}

__global__ void k_nchw_to_tok(const float* __restrict__ x, float* __restrict__ tok,
                              int F, int C, int H, int W)
{
  int i = blockIdx.x * blockDim.x + threadIdx.x;
  int total = F * C * H * W;
  if (i >= total) return;
  int c = i % C; int p = i / C;
  int xx = p % W, yy = (p / W) % H, f = p / (W * H);
  tok[i] = x[(((size_t)f * C + c) * H + yy) * W + xx];
}

__global__ void k_tok_to_nchw(const float* __restrict__ tok, float* __restrict__ x,
                              int F, int C, int H, int W)
{
  int i = blockIdx.x * blockDim.x + threadIdx.x;
  int total = F * C * H * W;
  if (i >= total) return;
  int xx = i % W, yy = (i / W) % H, c = (i / (W * H)) % C, f = i / (W * H * C);
  x[i] = tok[((size_t)((f * H + yy) * W + xx)) * C + c];
}

__global__ void k_win_gather(const float* __restrict__ tok, float* __restrict__ xw,
                             int shifted)
{
  int i = blockIdx.x * blockDim.x + threadIdx.x;
  int total = TOKP * EMBED;
  if (i >= total) return;
  int c = i % EMBED; int p = i / EMBED;
  int l = p & 127, wi = p >> 7;
  int zx = l & 7, zy = (l >> 3) & 7, zd = l >> 6;
  int bx = wi & 7, by = (wi >> 3) & 7, bd = wi >> 6;
  int d = bd * 2 + zd, y = by * 8 + zy, x = bx * 8 + zx;
  if (shifted) { d = (d + 1) & 3; y = (y + 4) & 63; x = (x + 4) & 63; }
  float v = 0.f;
  if (d < 3) v = tok[((size_t)((d * HH + y) * WW + x)) * EMBED + c];
  xw[i] = v;
}

__global__ void k_win_scatter(const float* __restrict__ win, const float* __restrict__ res,
                              float* __restrict__ y, int shifted)
{
  int i = blockIdx.x * blockDim.x + threadIdx.x;
  int total = TOK3 * EMBED;
  if (i >= total) return;
  int c = i % EMBED; int p = i / EMBED;
  int x = p % WW, yy = (p / WW) % HH, d = p / (WW * HH);
  int dc = d, yc = yy, xc = x;
  if (shifted) { dc = (d + 3) & 3; yc = (yy + 60) & 63; xc = (x + 60) & 63; }
  int bd = dc >> 1, zd = dc & 1;
  int by = yc >> 3, zy = yc & 7;
  int bx = xc >> 3, zx = xc & 7;
  int wi = (bd * 8 + by) * 8 + bx;
  int l  = (zd * 8 + zy) * 8 + zx;
  y[i] = res[i] + win[((size_t)(wi * 128 + l)) * EMBED + c];
}

__global__ void k_iw_prep(const float* __restrict__ supp, const float* __restrict__ flow,
                          float* __restrict__ kv, float* __restrict__ kvpe)
{
  int i = blockIdx.x * blockDim.x + threadIdx.x;
  int total = HWX * 4 * EMBED;
  if (i >= total) return;
  int c = i % EMBED;
  int k = (i / EMBED) & 3;
  int p = i / (EMBED * 4);
  int y = p / WW, x = p % WW;
  float px = x + flow[p];
  float py = y + flow[HWX + p];
  float x0 = floorf(px), y0 = floorf(py);
  float fx = px - x0, fy = py - y0;
  int oy = k >> 1, ox = k & 1;
  int ny = (int)(y0) + oy; ny = ny < 0 ? 0 : (ny > HH - 1 ? HH - 1 : ny);
  int nx = (int)(x0) + ox; nx = nx < 0 ? 0 : (nx > WW - 1 ? WW - 1 : nx);
  float val = supp[((size_t)c * HH + ny) * WW + nx];
  float dy = (float)oy - fy, dx = (float)ox - fx;
  int j = c % 20, grp = c / 20;
  float freq = __powf(0.01f, j * 0.05f);
  float pe;
  if (grp == 0)      pe = __sinf(dy / freq);
  else if (grp == 1) pe = __cosf(dy / freq);
  else if (grp == 2) pe = __sinf(dx / freq);
  else               pe = __cosf(dx / freq);
  kv[i] = val;
  kvpe[i] = val + pe;
}

__global__ void k_iw_attn(const float* __restrict__ q, const float* __restrict__ k,
                          const float* __restrict__ v, float* __restrict__ o)
{
  int t = blockIdx.x * blockDim.x + threadIdx.x;
  if (t >= HWX * HEADS) return;
  int pix = t / HEADS, hd = t % HEADS;
  const float* qr = q + (size_t)pix * EMBED + hd * DH;
  float lg[4];
  float mx = -3.4e38f;
  for (int kk = 0; kk < 4; ++kk) {
    const float* kr = k + ((size_t)(pix * 4 + kk)) * EMBED + hd * DH;
    float s = 0.f;
    for (int j = 0; j < DH; ++j) s += qr[j] * kr[j];
    s *= ATT_SCALE;
    lg[kk] = s;
    mx = fmaxf(mx, s);
  }
  float sum = 0.f;
  for (int kk = 0; kk < 4; ++kk) { lg[kk] = __expf(lg[kk] - mx); sum += lg[kk]; }
  float inv = 1.f / sum;
  float* orow = o + (size_t)pix * EMBED + hd * DH;
  for (int j = 0; j < DH; ++j) {
    float s = 0.f;
    for (int kk = 0; kk < 4; ++kk)
      s += lg[kk] * v[((size_t)(pix * 4 + kk)) * EMBED + hd * DH + j];
    orow[j] = s * inv;
  }
}

__global__ void k_warp_add_flow(const float* __restrict__ n1, const float* __restrict__ n2,
                                float* __restrict__ out)
{
  int i = blockIdx.x * blockDim.x + threadIdx.x;
  if (i >= 2 * HWX) return;
  int c = i / HWX, p = i % HWX;
  int y = p / WW, x = p % WW;
  float px = x + n1[p], py = y + n1[HWX + p];
  float x0f = floorf(px), y0f = floorf(py);
  float wx = px - x0f, wy = py - y0f;
  int x0 = (int)x0f, y0 = (int)y0f;
  int x0c = x0 < 0 ? 0 : (x0 > WW - 1 ? WW - 1 : x0);
  int x1c = x0 + 1 < 0 ? 0 : (x0 + 1 > WW - 1 ? WW - 1 : x0 + 1);
  int y0c = y0 < 0 ? 0 : (y0 > HH - 1 ? HH - 1 : y0);
  int y1c = y0 + 1 < 0 ? 0 : (y0 + 1 > HH - 1 ? HH - 1 : y0 + 1);
  const float* img = n2 + (size_t)c * HWX;
  float v = img[y0c * WW + x0c] * (1 - wx) * (1 - wy) +
            img[y0c * WW + x1c] * wx * (1 - wy) +
            img[y1c * WW + x0c] * (1 - wx) * wy +
            img[y1c * WW + x1c] * wx * wy;
  out[i] = n1[i] + v;
}

__global__ void k_extract_add(const float* __restrict__ bb, const float* __restrict__ curr,
                              float* __restrict__ dst)
{
  int i = blockIdx.x * blockDim.x + threadIdx.x;
  if (i >= FS) return;
  dst[i] = bb[FS + i] + curr[i];
}

__global__ void k_pixel_shuffle(const float* __restrict__ x, float* __restrict__ y,
                                int N, int C, int H, int W, int r)
{
  int i = blockIdx.x * blockDim.x + threadIdx.x;
  int Co = C / (r * r);
  int total = N * Co * H * r * W * r;
  if (i >= total) return;
  int Wr = W * r, Hr = H * r;
  int ox = i % Wr, oy = (i / Wr) % Hr, co = (i / (Wr * Hr)) % Co, n = i / (Wr * Hr * Co);
  int r1 = oy % r, r2 = ox % r, ys = oy / r, xs = ox / r;
  int ci = co * r * r + r1 * r + r2;
  y[i] = x[(((size_t)n * C + ci) * H + ys) * W + xs];
}

__global__ void k_leaky(float* p, int n)
{
  int i = blockIdx.x * blockDim.x + threadIdx.x;
  if (i >= n) return;
  float v = p[i];
  p[i] = (v >= 0.f) ? v : 0.1f * v;
}

__global__ void k_up4_add(const float* __restrict__ lq, float* __restrict__ y,
                          int C, int H, int W)
{
  int i = blockIdx.x * blockDim.x + threadIdx.x;
  int total = C * 4 * H * 4 * W;
  if (i >= total) return;
  int W4 = 4 * W, H4 = 4 * H;
  int ox = i % W4, oy = (i / W4) % H4, c = i / (W4 * H4);
  float sx = (ox + 0.5f) * 0.25f - 0.5f;
  float sy = (oy + 0.5f) * 0.25f - 0.5f;
  int x0 = (int)floorf(sx), y0 = (int)floorf(sy);
  float wx = sx - x0, wy = sy - y0;
  int x0c = x0 < 0 ? 0 : (x0 > W - 1 ? W - 1 : x0);
  int x1c = x0 + 1 < 0 ? 0 : (x0 + 1 > W - 1 ? W - 1 : x0 + 1);
  int y0c = y0 < 0 ? 0 : (y0 > H - 1 ? H - 1 : y0);
  int y1c = y0 + 1 < 0 ? 0 : (y0 + 1 > H - 1 ? H - 1 : y0 + 1);
  const float* img = lq + (size_t)c * H * W;
  float v = img[y0c * W + x0c] * (1 - wx) * (1 - wy) +
            img[y0c * W + x1c] * wx * (1 - wy) +
            img[y1c * W + x0c] * (1 - wx) * wy +
            img[y1c * W + x1c] * wx * wy;
  y[i] += v;
}

// ---------------------------------------------------------------------------
// Host-side orchestration
// ---------------------------------------------------------------------------
struct BlockP {
  const float *fc1_b, *fc1_w, *fc2_b, *fc2_w, *ln1_b, *ln1_g, *ln2_b, *ln2_g,
              *proj_b, *proj_w, *qkv_b, *qkv_w;
};
struct FexP { BlockP blk[9]; const float *epi_b, *epi_w, *pre_b, *pre_w; };
struct IwP  { const float *wq, *wk, *wv, *wo; };

struct BlockW { const _Float16 *qkv, *proj, *fc1, *fc2; };
struct FexW { BlockW blk[9]; const _Float16 *pre, *epi; };
struct IwW  { const _Float16 *wq, *wk, *wv, *wo; };

struct WS {
  float *featsA, *featsB;
  float *cond, *convy, *z, *z2, *ln, *xw, *qkv, *attno, *projo, *h1, *znchw, *bbout;
  float *kv, *kvpe, *tokcur, *q, *kk, *vv, *ao, *o, *n1c, *n2c, *n2flow;
  float *ux, *u1, *p1, *u2, *p2, *hr;
};

static inline int gdiv(long n) { return (int)((n + 255) / 256); }

static void gemm(hipStream_t st, const float* X, const _Float16* W, const float* bias,
                 const float* res, float* Y, int M, int N, int K, int act)
{
  dim3 g((N + 63) / 64, (M + 63) / 64);
  k_gemm<<<g, 256, 0, st>>>(X, W, bias, res, Y, M, N, K, act);
}

static void convg(hipStream_t st, const float* x, const _Float16* wpk, const float* b,
                  float* y, int NB, int Ci, int Co, int H, int W, int K, int pad, int act)
{
  long Mrows = (long)NB * H * W;
  dim3 g((Co + 63) / 64, (unsigned)((Mrows + 63) / 64));
  k_conv_gemm<<<g, 256, 0, st>>>(x, wpk, b, y, NB, Ci, Co, H, W, K, pad, act);
}

static void backbone(hipStream_t st, WS& w, const FexP& fp, const FexW& fw)
{
  convg(st, w.cond, fw.pre, fp.pre_b, w.convy, 3, EMBED, EMBED, HH, WW, 7, 3, 0);
  k_nchw_to_tok<<<gdiv((long)3 * EMBED * HWX), 256, 0, st>>>(w.convy, w.z, 3, EMBED, HH, WW);
  for (int b = 0; b < 9; ++b) {
    const BlockP& bp = fp.blk[b];
    const BlockW& bw = fw.blk[b];
    int sh = b & 1;
    k_layernorm<<<gdiv(TOK3), 256, 0, st>>>(w.z, bp.ln1_g, bp.ln1_b, w.ln, TOK3);
    k_win_gather<<<gdiv((long)TOKP * EMBED), 256, 0, st>>>(w.ln, w.xw, sh);
    gemm(st, w.xw, bw.qkv, bp.qkv_b, nullptr, w.qkv, TOKP, 3 * EMBED, EMBED, 0);
    k_win_attn<<<dim3(128 * HEADS), 256, 0, st>>>(w.qkv, w.attno, 128);
    gemm(st, w.attno, bw.proj, bp.proj_b, nullptr, w.projo, TOKP, EMBED, EMBED, 0);
    k_win_scatter<<<gdiv((long)TOK3 * EMBED), 256, 0, st>>>(w.projo, w.z, w.z2, sh);
    k_layernorm<<<gdiv(TOK3), 256, 0, st>>>(w.z2, bp.ln2_g, bp.ln2_b, w.ln, TOK3);
    gemm(st, w.ln, bw.fc1, bp.fc1_b, nullptr, w.h1, TOK3, 4 * EMBED, EMBED, 1);
    gemm(st, w.h1, bw.fc2, bp.fc2_b, w.z2, w.z, TOK3, EMBED, 4 * EMBED, 0);
  }
  k_tok_to_nchw<<<gdiv((long)3 * EMBED * HWX), 256, 0, st>>>(w.z, w.znchw, 3, EMBED, HH, WW);
  convg(st, w.znchw, fw.epi, fp.epi_b, w.bbout, 3, EMBED, EMBED, HH, WW, 3, 1, 0);
}

static void implicit_warp(hipStream_t st, WS& w, const IwW& iw,
                          const float* supp, const float* curr, const float* flow,
                          float* dst)
{
  k_iw_prep<<<gdiv((long)HWX * 4 * EMBED), 256, 0, st>>>(supp, flow, w.kv, w.kvpe);
  k_nchw_to_tok<<<gdiv((long)FS), 256, 0, st>>>(curr, w.tokcur, 1, EMBED, HH, WW);
  gemm(st, w.tokcur, iw.wq, nullptr, nullptr, w.q, HWX, EMBED, EMBED, 0);
  gemm(st, w.kvpe, iw.wk, nullptr, nullptr, w.kk, 4 * HWX, EMBED, EMBED, 0);
  gemm(st, w.kv,   iw.wv, nullptr, nullptr, w.vv, 4 * HWX, EMBED, EMBED, 0);
  k_iw_attn<<<gdiv((long)HWX * HEADS), 256, 0, st>>>(w.q, w.kk, w.vv, w.ao);
  gemm(st, w.ao, iw.wo, nullptr, nullptr, w.o, HWX, EMBED, EMBED, 0);
  k_tok_to_nchw<<<gdiv((long)FS), 256, 0, st>>>(w.o, dst, 1, EMBED, HH, WW);
}

static void propagate(hipStream_t st, WS& w, const float* fin, float* fout,
                      const float* flows, const FexP& fp, const FexW& fw,
                      const IwW& iw, int rev)
{
  const size_t FB = FS * sizeof(float);
  for (int i = 0; i < 6; ++i) {
    int fr = rev ? i : 5 - i;
    int slot = rev ? 5 - i : i;
    const float* curr = fin + (size_t)fr * FS;
    const float* n1cond = curr;
    const float* n2cond = curr;
    if (i > 0) {
      int f1 = rev ? (i - 1) : (5 - i);
      const float* n1f = flows + (size_t)f1 * FLS;
      int s1 = rev ? 5 - (i - 1) : (i - 1);
      implicit_warp(st, w, iw, fout + (size_t)s1 * FS, curr, n1f, w.n1c);
      n1cond = w.n1c;
      if (i > 1) {
        int f2 = rev ? (i - 2) : (6 - i);
        k_warp_add_flow<<<gdiv(2 * HWX), 256, 0, st>>>(n1f, flows + (size_t)f2 * FLS,
                                                       w.n2flow);
        int s2 = rev ? 5 - (i - 2) : (i - 2);
        implicit_warp(st, w, iw, fout + (size_t)s2 * FS, curr, w.n2flow, w.n2c);
        n2cond = w.n2c;
      }
    }
    hipMemcpyAsync(w.cond,          n1cond, FB, hipMemcpyDeviceToDevice, st);
    hipMemcpyAsync(w.cond + FS,     curr,   FB, hipMemcpyDeviceToDevice, st);
    hipMemcpyAsync(w.cond + 2 * FS, n2cond, FB, hipMemcpyDeviceToDevice, st);
    backbone(st, w, fp, fw);
    k_extract_add<<<gdiv(FS), 256, 0, st>>>(w.bbout, curr, fout + (size_t)slot * FS);
  }
}

extern "C" void kernel_launch(void* const* d_in, const int* in_sizes, int n_in,
                              void* d_out, int out_size, void* d_ws, size_t ws_size,
                              hipStream_t stream)
{
  (void)in_sizes; (void)n_in; (void)out_size; (void)ws_size;
  const float* lqs = (const float*)d_in[0];
  const float* fwd = (const float*)d_in[1];
  const float* bwd = (const float*)d_in[2];
  int pi = 3;
  auto F = [&](int i) { return (const float*)d_in[i]; };
  const float* cbu_b = F(pi++); const float* cbu_w = F(pi++);
  FexP fex[4];
  for (int s = 0; s < 4; ++s) {
    for (int b = 0; b < 9; ++b) {
      BlockP& bp = fex[s].blk[b];
      bp.fc1_b = F(pi++); bp.fc1_w = F(pi++); bp.fc2_b = F(pi++); bp.fc2_w = F(pi++);
      bp.ln1_b = F(pi++); bp.ln1_g = F(pi++); bp.ln2_b = F(pi++); bp.ln2_g = F(pi++);
      bp.proj_b = F(pi++); bp.proj_w = F(pi++); bp.qkv_b = F(pi++); bp.qkv_w = F(pi++);
    }
    fex[s].epi_b = F(pi++); fex[s].epi_w = F(pi++);
    fex[s].pre_b = F(pi++); fex[s].pre_w = F(pi++);
  }
  const float* hr_b = F(pi++); const float* hr_w = F(pi++);
  IwP iwp; iwp.wk = F(pi++); iwp.wo = F(pi++); iwp.wq = F(pi++); iwp.wv = F(pi++);
  const float* last_b = F(pi++); const float* last_w = F(pi++);
  const float* sp_b = F(pi++); const float* sp_w = F(pi++);
  const float* up1_b = F(pi++); const float* up1_w = F(pi++);
  const float* up2_b = F(pi++); const float* up2_w = F(pi++);

  float* base = (float*)d_ws;
  size_t off = 0;
  auto alloc  = [&](size_t n) { float* p = base + off; off += n; return p; };
  auto allocH = [&](size_t n) { _Float16* p = (_Float16*)(base + off);
                                off += (n + 1) / 2; return p; };

  WS w;
  w.featsA = alloc(6 * (size_t)FS);
  w.featsB = alloc(6 * (size_t)FS);
  w.cond   = alloc(3 * (size_t)FS);
  w.convy  = alloc(3 * (size_t)FS);
  w.z      = alloc(3 * (size_t)FS);
  w.z2     = alloc(3 * (size_t)FS);
  w.ln     = alloc(3 * (size_t)FS);
  w.xw     = alloc((size_t)TOKP * EMBED);
  w.qkv    = alloc((size_t)TOKP * 3 * EMBED);
  w.attno  = alloc((size_t)TOKP * EMBED);
  w.projo  = alloc((size_t)TOKP * EMBED);
  w.h1     = alloc((size_t)TOK3 * 4 * EMBED);
  w.znchw  = alloc(3 * (size_t)FS);
  w.bbout  = alloc(3 * (size_t)FS);
  w.kv     = alloc(4 * (size_t)FS);
  w.kvpe   = alloc(4 * (size_t)FS);
  w.tokcur = alloc((size_t)FS);
  w.q      = alloc((size_t)FS);
  w.kk     = alloc(4 * (size_t)FS);
  w.vv     = alloc(4 * (size_t)FS);
  w.ao     = alloc((size_t)FS);
  w.o      = alloc((size_t)FS);
  w.n1c    = alloc((size_t)FS);
  w.n2c    = alloc((size_t)FS);
  w.n2flow = alloc((size_t)FLS);
  w.ux     = alloc((size_t)32 * HWX);
  w.u1     = alloc((size_t)128 * HWX);
  w.p1     = alloc((size_t)32 * 4 * HWX);
  w.u2     = alloc((size_t)256 * 4 * HWX);
  w.p2     = alloc((size_t)64 * 16 * HWX);
  w.hr     = alloc((size_t)64 * 16 * HWX);

  // ---- pack all matrix weights to f16 panels (idempotent, graph-safe)
  auto packW = [&](const float* src, int N, int K) {
    _Float16* d = allocH((size_t)N * K);
    k_pack_w<<<gdiv((long)N * K), 256, 0, stream>>>(src, d, N * K);
    return (const _Float16*)d;
  };
  auto packWT = [&](const float* src, int N, int K) {
    _Float16* d = allocH((size_t)N * K);
    k_pack_wT<<<gdiv((long)N * K), 256, 0, stream>>>(src, d, N, K);
    return (const _Float16*)d;
  };
  auto packC = [&](const float* src, int Co, int Ci, int Kk) {
    _Float16* d = allocH((size_t)Kk * Kk * Co * Ci);
    k_pack_conv<<<gdiv((long)Kk * Kk * Co * Ci), 256, 0, stream>>>(src, d, Co, Ci, Kk);
    return (const _Float16*)d;
  };

  FexW fexw[4];
  for (int s = 0; s < 4; ++s) {
    for (int b = 0; b < 9; ++b) {
      const BlockP& bp = fex[s].blk[b];
      BlockW& bw = fexw[s].blk[b];
      bw.qkv  = packW(bp.qkv_w, 3 * EMBED, EMBED);
      bw.proj = packW(bp.proj_w, EMBED, EMBED);
      bw.fc1  = packW(bp.fc1_w, 4 * EMBED, EMBED);
      bw.fc2  = packW(bp.fc2_w, EMBED, 4 * EMBED);
    }
    fexw[s].pre = packC(fex[s].pre_w, EMBED, EMBED, 7);
    fexw[s].epi = packC(fex[s].epi_w, EMBED, EMBED, 3);
  }
  IwW iww;
  iww.wq = packWT(iwp.wq, EMBED, EMBED);
  iww.wk = packWT(iwp.wk, EMBED, EMBED);
  iww.wv = packWT(iwp.wv, EMBED, EMBED);
  iww.wo = packWT(iwp.wo, EMBED, EMBED);
  const _Float16* spw  = packC(sp_w, EMBED, 3, 3);
  const _Float16* cbuw = packC(cbu_w, 32, EMBED, 3);
  const _Float16* u1w  = packC(up1_w, 128, 32, 3);
  const _Float16* u2w  = packC(up2_w, 256, 32, 3);
  const _Float16* hrw  = packC(hr_w, 64, 64, 3);
  const _Float16* lstw = packC(last_w, 3, 64, 3);

  // ---- spatial embedding
  convg(stream, lqs, spw, sp_b, w.featsA, 6, 3, EMBED, HH, WW, 3, 1, 0);

  // ---- 4 propagation passes
  propagate(stream, w, w.featsA, w.featsB, fwd, fex[0], fexw[0], iww, 0);
  propagate(stream, w, w.featsB, w.featsA, bwd, fex[1], fexw[1], iww, 1);
  propagate(stream, w, w.featsA, w.featsB, fwd, fex[2], fexw[2], iww, 0);
  propagate(stream, w, w.featsB, w.featsA, bwd, fex[3], fexw[3], iww, 1);

  // ---- upsampler, per frame
  for (int t = 0; t < 6; ++t) {
    const float* ft = w.featsA + (size_t)t * FS;
    convg(stream, ft, cbuw, cbu_b, w.ux, 1, EMBED, 32, HH, WW, 3, 1, 0);
    convg(stream, w.ux, u1w, up1_b, w.u1, 1, 32, 128, HH, WW, 3, 1, 0);
    k_pixel_shuffle<<<gdiv((long)32 * 4 * HWX), 256, 0, stream>>>(w.u1, w.p1, 1, 128, HH, WW, 2);
    k_leaky<<<gdiv((long)32 * 4 * HWX), 256, 0, stream>>>(w.p1, 32 * 4 * HWX);
    convg(stream, w.p1, u2w, up2_b, w.u2, 1, 32, 256, 2 * HH, 2 * WW, 3, 1, 0);
    k_pixel_shuffle<<<gdiv((long)64 * 16 * HWX), 256, 0, stream>>>(w.u2, w.p2, 1, 256, 2 * HH, 2 * WW, 2);
    k_leaky<<<gdiv((long)64 * 16 * HWX), 256, 0, stream>>>(w.p2, 64 * 16 * HWX);
    convg(stream, w.p2, hrw, hr_b, w.hr, 1, 64, 64, 4 * HH, 4 * WW, 3, 1, 1);
    float* outf = (float*)d_out + (size_t)t * 3 * 16 * HWX;
    convg(stream, w.hr, lstw, last_b, outf, 1, 64, 3, 4 * HH, 4 * WW, 3, 1, 0);
    k_up4_add<<<gdiv((long)3 * 16 * HWX), 256, 0, stream>>>(lqs + (size_t)t * 3 * HWX,
                                                            outf, 3, HH, WW);
  }
}
